// RoastLinear_21603685499740
// MI455X (gfx1250) — matrix operand order repaired and checked by reading, not compiled, not run on hardware
//
#include <hip/hip_runtime.h>

#define IDIM  2048
#define ODIM  8192
#define BATCH 4096

typedef __attribute__((ext_vector_type(16))) __bf16 v16bf;
typedef __attribute__((ext_vector_type(8)))  __bf16 v8bf;
typedef __attribute__((ext_vector_type(8)))  float  v8f;
typedef __attribute__((ext_vector_type(4)))  int    v4i;

// ---------------------------------------------------------------------------
// Pass 1a: convert x (f32 [BATCH][IDIM]) -> xb (bf16, row-major)
// ---------------------------------------------------------------------------
__global__ __launch_bounds__(256) void cvt_x_kernel(const float* __restrict__ x,
                                                    __bf16* __restrict__ xb) {
  size_t i = ((size_t)blockIdx.x * blockDim.x + threadIdx.x) * 8;
  v8f a = *(const v8f*)(x + i);
  v8bf o = __builtin_convertvector(a, v8bf);
  *(v8bf*)(xb + i) = o;
}

// ---------------------------------------------------------------------------
// Pass 1b: gather W[i,o] = weight[IDX[i,o]] * G[i,o], write transposed as
// Wt[o][i] in bf16. LDS transpose keeps IDX/G reads and Wt writes coalesced.
// ---------------------------------------------------------------------------
__global__ __launch_bounds__(256) void gather_w_kernel(
    const float* __restrict__ weight, const long long* __restrict__ idx,
    const float* __restrict__ g, __bf16* __restrict__ wt) {
  __shared__ __bf16 tile[64][72];
  const int i0 = blockIdx.x * 64;  // over IDIM
  const int o0 = blockIdx.y * 64;  // over ODIM
  const int t = threadIdx.x;

  const int ol = t & 63;
  const int ib = (t >> 6) * 16;
#pragma unroll 4
  for (int r = 0; r < 16; ++r) {
    const int il = ib + r;
    const size_t p = (size_t)(i0 + il) * ODIM + (o0 + ol);
    const long long id = idx[p];
    tile[il][ol] = (__bf16)(weight[id] * g[p]);
  }
  __syncthreads();

  const int il2 = t & 63;
  const int ob = (t >> 6) * 16;
#pragma unroll 4
  for (int r = 0; r < 16; ++r) {
    const int ol2 = ob + r;
    wt[(size_t)(o0 + ol2) * IDIM + (i0 + il2)] = tile[il2][ol2];
  }
}

// ---------------------------------------------------------------------------
// Pass 2: GEMM  out = xb @ Wt^T + bias   (bf16 inputs, f32 accumulate)
// Block tile 128x128x64, 256 threads (8 wave32), wave tile 64x32.
// Double-buffered LDS fed by GLOBAL_LOAD_ASYNC_TO_LDS_B128 when available.
// ---------------------------------------------------------------------------
#define BM 128
#define BN 128
#define BK 64
#define KPAD 8            // +16B/row: keeps 16B alignment, breaks bank conflicts
#define LDA (BK + KPAD)   // 72 halfs = 144B row stride
#define KTILES (IDIM / BK)

#if defined(__AMDGCN__) && \
    __has_builtin(__builtin_amdgcn_global_load_async_to_lds_b128) && \
    __has_builtin(__builtin_amdgcn_s_wait_asynccnt)
#define USE_ASYNC_LDS 1
#endif

#ifdef USE_ASYNC_LDS
typedef __attribute__((address_space(1))) v4i gbl_v4i;
typedef __attribute__((address_space(3))) v4i lds_v4i;
__device__ __forceinline__ void async_ld_b128(const void* g, void* l) {
  __builtin_amdgcn_global_load_async_to_lds_b128((gbl_v4i*)g, (lds_v4i*)l,
                                                 /*offset=*/0, /*cpol=*/0);
}
#endif

__global__ __launch_bounds__(256) void gemm_kernel(
    const __bf16* __restrict__ xb, const __bf16* __restrict__ wt,
    const float* __restrict__ bias, float* __restrict__ out) {
#ifdef USE_ASYNC_LDS
  __shared__ __bf16 As[2][BM][LDA];
  __shared__ __bf16 Bs[2][BN][LDA];
#else
  __shared__ __bf16 As[1][BM][LDA];
  __shared__ __bf16 Bs[1][BN][LDA];
#endif

  const int bn0 = blockIdx.x * BN;
  const int bm0 = blockIdx.y * BM;
  const int tid = threadIdx.x;
  const int lane = tid & 31;
  const int wid = tid >> 5;
  const int m0w = (wid & 1) * 64;   // wave M offset inside block
  const int n0w = (wid >> 1) * 32;  // wave N offset inside block
  const int ln16 = lane & 15;
  const int lhi = lane >> 4;

  v8f acc[4][2] = {};

  // staging map: 1024 16B-chunks per 128x64 tile; 4 chunks/thread/operand
  const int srow = tid >> 3;            // 0..31 base row group
  const int scol = (tid & 7) * 8;       // half offset 0..56

#ifdef USE_ASYNC_LDS
  // -------- async double-buffered pipeline --------
  // prologue: stage tile 0 into buffer 0
#pragma unroll
  for (int u = 0; u < 4; ++u) {
    const int row = srow + u * 32;
    async_ld_b128(xb + (size_t)(bm0 + row) * IDIM + scol, &As[0][row][scol]);
    async_ld_b128(wt + (size_t)(bn0 + row) * IDIM + scol, &Bs[0][row][scol]);
  }

  for (int kt = 0; kt < KTILES; ++kt) {
    const int cur = kt & 1;
    __builtin_amdgcn_s_wait_asynccnt(0);  // my async loads of tile kt landed
    __syncthreads();                      // everyone's landed; prev reads done

    if (kt + 1 < KTILES) {                // stage tile kt+1 into other buffer
      const size_t koff = (size_t)(kt + 1) * BK;
#pragma unroll
      for (int u = 0; u < 4; ++u) {
        const int row = srow + u * 32;
        async_ld_b128(xb + (size_t)(bm0 + row) * IDIM + koff + scol,
                      &As[cur ^ 1][row][scol]);
        async_ld_b128(wt + (size_t)(bn0 + row) * IDIM + koff + scol,
                      &Bs[cur ^ 1][row][scol]);
      }
    }
#else
  for (int kt = 0; kt < KTILES; ++kt) {
    const int cur = 0;
    const size_t koff = (size_t)kt * BK;
#pragma unroll
    for (int u = 0; u < 4; ++u) {
      const int row = srow + u * 32;
      *(v8bf*)&As[0][row][scol] =
          *(const v8bf*)(xb + (size_t)(bm0 + row) * IDIM + koff + scol);
      *(v8bf*)&Bs[0][row][scol] =
          *(const v8bf*)(wt + (size_t)(bn0 + row) * IDIM + koff + scol);
    }
    __syncthreads();
#endif

    // -------- compute: 2 K-substeps of 32, 8 WMMAs each --------
#pragma unroll
    for (int ks = 0; ks < 2; ++ks) {
      const int kb0 = ks * 32;
      // A (16x32): lanes 0-15 row M=lane, K 0-7 (v0-3) & 16-23 (v4-7);
      //            lanes 16-31 same M, K 8-15 & 24-31.
      v16bf af[4];
#pragma unroll
      for (int mi = 0; mi < 4; ++mi) {
        const int r = m0w + mi * 16 + ln16;
        const int kb = kb0 + lhi * 8;
        union { v16bf v; v8bf h[2]; } u;
        u.h[0] = *(const v8bf*)&As[cur][r][kb];
        u.h[1] = *(const v8bf*)&As[cur][r][kb + 16];
        af[mi] = u.v;
      }
      // B (32x16): lanes 0-15 col N=lane, K 0-7 & 16-23; lanes 16-31 K 8-15 & 24-31.
      v16bf bfr[2];
#pragma unroll
      for (int ni = 0; ni < 2; ++ni) {
        const int c = n0w + ni * 16 + ln16;
        const int kb = kb0 + lhi * 8;
        union { v16bf v; v8bf h[2]; } u;
        u.h[0] = *(const v8bf*)&Bs[cur][c][kb];
        u.h[1] = *(const v8bf*)&Bs[cur][c][kb + 16];
        bfr[ni] = u.v;
      }
#pragma unroll
      for (int mi = 0; mi < 4; ++mi)
#pragma unroll
        for (int ni = 0; ni < 2; ++ni)
          acc[mi][ni] = __builtin_amdgcn_wmma_f32_16x16x32_bf16(
              false, af[mi], false, bfr[ni], (short)0, acc[mi][ni], false,
              false);
    }
#ifndef USE_ASYNC_LDS
    __syncthreads();
#endif
  }

  // -------- epilogue: C/D layout: VGPR r -> M = r (lanes 0-15) or r+8,
  //          N = lane%16 --------
#pragma unroll
  for (int mi = 0; mi < 4; ++mi) {
#pragma unroll
    for (int ni = 0; ni < 2; ++ni) {
      const int n = bn0 + n0w + ni * 16 + ln16;
      const float bv = bias[n];
#pragma unroll
      for (int r = 0; r < 8; ++r) {
        const int m = bm0 + m0w + mi * 16 + lhi * 8 + r;
        out[(size_t)m * ODIM + n] = acc[mi][ni][r] + bv;
      }
    }
  }
}

// ---------------------------------------------------------------------------
extern "C" void kernel_launch(void* const* d_in, const int* in_sizes, int n_in,
                              void* d_out, int out_size, void* d_ws,
                              size_t ws_size, hipStream_t stream) {
  (void)in_sizes; (void)n_in; (void)out_size; (void)ws_size;
  const float* x       = (const float*)d_in[0];
  const float* weight  = (const float*)d_in[1];
  const long long* IDX = (const long long*)d_in[2];  // int64 per reference
  const float* G       = (const float*)d_in[3];
  const float* bias    = (const float*)d_in[4];
  float* out           = (float*)d_out;

  // workspace: xb bf16 [BATCH][IDIM] (16MB) | Wt bf16 [ODIM][IDIM] (32MB)
  __bf16* xb = (__bf16*)d_ws;
  __bf16* wt = xb + (size_t)BATCH * IDIM;

  cvt_x_kernel<<<(BATCH * IDIM / 8) / 256, 256, 0, stream>>>(x, xb);

  dim3 gg(IDIM / 64, ODIM / 64);
  gather_w_kernel<<<gg, 256, 0, stream>>>(weight, IDX, G, wt);

  dim3 g3(ODIM / BN, BATCH / BM);
  gemm_kernel<<<g3, 256, 0, stream>>>(xb, wt, bias, out);
}
